// SiameseGRU_53412213293757
// MI455X (gfx1250) — compile-verified, matrix operand order checked by reading, not run on hardware
//
#include <hip/hip_runtime.h>

#define T_LEN 1024
#define HSTR  72   // padded LDS row stride (halves): 144B rows, 16B aligned, bank-conflict-free

typedef __attribute__((ext_vector_type(16))) _Float16 v16h;
typedef __attribute__((ext_vector_type(8)))  _Float16 v8h;
typedef __attribute__((ext_vector_type(8)))  float    v8f;

// Fast activations: v_exp_f32 + v_rcp_f32 (TRANS ops, co-execute with WMMA pipe).
// NOTE: __frcp_rn lowers to the full IEEE div sequence (v_div_scale/_fmas/_fixup);
// __builtin_amdgcn_rcpf is the single-instruction hardware reciprocal.
__device__ __forceinline__ float fast_sigmoid(float x) {
    return __builtin_amdgcn_rcpf(1.0f + __expf(-x));
}
__device__ __forceinline__ float fast_tanh(float x) {
    return 2.0f * __builtin_amdgcn_rcpf(1.0f + __expf(-2.0f * x)) - 1.0f;
}

__device__ __forceinline__ v16h make_frag16(v8h lo, v8h hi) {
    return __builtin_shufflevector(lo, hi, 0,1,2,3,4,5,6,7,8,9,10,11,12,13,14,15);
}

// A-fragment (16x32 f16, MxK). ISA layout: lanes 0-15 hold M=lane, K = k0+{0..7, 16..23};
// lanes 16-31 hold M=lane-16, K = k0+{8..15, 24..31}. Two 16B LDS chunks per lane.
__device__ __forceinline__ v16h load_a_frag(const _Float16* hb, int lane, int k0) {
    const int m    = lane & 15;
    const int koff = k0 + ((lane >> 4) << 3);
    const _Float16* p = hb + m * HSTR + koff;
    v8h lo = *(const v8h*)(p);
    v8h hi = *(const v8h*)(p + 16);
    return make_frag16(lo, hi);
}

// B-fragment (32x16 f16, KxN) for output row `ng` of W_hh (row-major [192][64]):
// B[K][n] = W_hh[ng(n)][K]; same per-lane K chunking as A with N in place of M.
__device__ __forceinline__ v16h load_b_frag(const float* __restrict__ Whh, int ng, int lane, int k0) {
    const int koff = k0 + ((lane >> 4) << 3);
    const float* p = Whh + ng * 64 + koff;
    v16h b;
    #pragma unroll
    for (int i = 0; i < 8; ++i) {
        b[i]     = (_Float16)p[i];
        b[8 + i] = (_Float16)p[16 + i];
    }
    return b;
}

__global__ __launch_bounds__(128, 1)
void siamese_gru_wmma(const float* __restrict__ x1, const float* __restrict__ x2,
                      const float* __restrict__ Wih, const float* __restrict__ Whh,
                      const float* __restrict__ bih, const float* __restrict__ bhh,
                      const float* __restrict__ W1,  const float* __restrict__ b1,
                      const float* __restrict__ W2,  const float* __restrict__ b2,
                      float* __restrict__ out)
{
    // [seq][parity][row][col] h state in f16 for A-fragment building
    __shared__ __align__(16) _Float16 hsh[2][2][16][HSTR];
    __shared__ float xsh[2][2][16][4];   // [parity][seq][row][dim]
    __shared__ float dsh[16][68];        // |h1-h2| for the MLP head

    const int tid  = threadIdx.x;
    const int lane = tid & 31;
    const int wv   = tid >> 5;           // wave 0..3
    const int np   = lane & 15;
    const int hi16 = lane >> 4;
    const int b0   = blockIdx.x * 16;

    // wave wv + lane column np own hidden column j for all three gates
    const int j   = wv * 16 + np;
    const int Ngr = j, Ngz = 64 + j, Ngn = 128 + j;

    // per-lane constants (biases + input weights, D=3)
    const float bsr = bih[Ngr] + bhh[Ngr];
    const float bsz = bih[Ngz] + bhh[Ngz];
    const float bin = bih[Ngn];
    const float bhn = bhh[Ngn];
    const float wr0 = Wih[Ngr*3], wr1 = Wih[Ngr*3+1], wr2 = Wih[Ngr*3+2];
    const float wz0 = Wih[Ngz*3], wz1 = Wih[Ngz*3+1], wz2 = Wih[Ngz*3+2];
    const float wn0 = Wih[Ngn*3], wn1 = Wih[Ngn*3+1], wn2 = Wih[Ngn*3+2];

    // recurrent weight fragments: loaded once, pinned in VGPRs for all T steps
    const v16h br0 = load_b_frag(Whh, Ngr, lane, 0);
    const v16h br1 = load_b_frag(Whh, Ngr, lane, 32);
    const v16h bz0 = load_b_frag(Whh, Ngz, lane, 0);
    const v16h bz1 = load_b_frag(Whh, Ngz, lane, 32);
    const v16h bn0 = load_b_frag(Whh, Ngn, lane, 0);
    const v16h bn1 = load_b_frag(Whh, Ngn, lane, 32);

    // h0 = 0
    for (int i = tid; i < 16 * HSTR; i += 128) {
        (&hsh[0][0][0][0])[i] = (_Float16)0.0f;
        (&hsh[1][0][0][0])[i] = (_Float16)0.0f;
    }
    // stage x_0 for both sequences
    if (tid < 32) {
        const int s = tid >> 4, r = tid & 15;
        const float* xp = (s ? x2 : x1) + ((size_t)(b0 + r) * T_LEN) * 3;
        xsh[0][s][r][0] = xp[0]; xsh[0][s][r][1] = xp[1]; xsh[0][s][r][2] = xp[2];
    }

    float hreg[2][8];
    #pragma unroll
    for (int s = 0; s < 2; ++s) {
        #pragma unroll
        for (int v = 0; v < 8; ++v) hreg[s][v] = 0.0f;
    }

    __syncthreads();

    for (int t = 0; t < T_LEN; ++t) {
        const int cur = t & 1, nxt = cur ^ 1;

        // prefetch x_{t+1} into the other parity buffer (overlaps WMMAs)
        if (tid < 32) {
            const int s = tid >> 4, r = tid & 15;
            const int tt = (t + 1 < T_LEN) ? (t + 1) : t;
            const float* xp = (s ? x2 : x1) + ((size_t)(b0 + r) * T_LEN + tt) * 3;
            xsh[nxt][s][r][0] = xp[0]; xsh[nxt][s][r][1] = xp[1]; xsh[nxt][s][r][2] = xp[2];
        }

        #pragma unroll
        for (int s = 0; s < 2; ++s) {
            const _Float16* hb = &hsh[s][cur][0][0];
            const v16h a0 = load_a_frag(hb, lane, 0);
            const v16h a1 = load_a_frag(hb, lane, 32);

            v8f cr, cz, cn;
            float xn[8];
            #pragma unroll
            for (int v = 0; v < 8; ++v) {
                const int m = v + (hi16 << 3);   // C-layout row for this VGPR slot
                const float xv0 = xsh[cur][s][m][0];
                const float xv1 = xsh[cur][s][m][1];
                const float xv2 = xsh[cur][s][m][2];
                cr[v] = bsr + xv0 * wr0 + xv1 * wr1 + xv2 * wr2;  // xr + b_ir + b_hr
                cz[v] = bsz + xv0 * wz0 + xv1 * wz1 + xv2 * wz2;  // xz + b_iz + b_hz
                xn[v] = bin + xv0 * wn0 + xv1 * wn1 + xv2 * wn2;  // xn + b_in (kept separate)
                cn[v] = bhn;                                      // hn accumulates b_hn only
            }

            // hg = h @ W_hh^T : K=64 in two 16x16x32 f16 WMMAs per gate tile
            cr = __builtin_amdgcn_wmma_f32_16x16x32_f16(false, a0, false, br0, (short)0, cr, false, false);
            cr = __builtin_amdgcn_wmma_f32_16x16x32_f16(false, a1, false, br1, (short)0, cr, false, false);
            cz = __builtin_amdgcn_wmma_f32_16x16x32_f16(false, a0, false, bz0, (short)0, cz, false, false);
            cz = __builtin_amdgcn_wmma_f32_16x16x32_f16(false, a1, false, bz1, (short)0, cz, false, false);
            cn = __builtin_amdgcn_wmma_f32_16x16x32_f16(false, a0, false, bn0, (short)0, cn, false, false);
            cn = __builtin_amdgcn_wmma_f32_16x16x32_f16(false, a1, false, bn1, (short)0, cn, false, false);

            #pragma unroll
            for (int v = 0; v < 8; ++v) {
                const float r = fast_sigmoid(cr[v]);
                const float z = fast_sigmoid(cz[v]);
                const float n = fast_tanh(xn[v] + r * cn[v]);      // PyTorch: r gates (hn + b_hn)
                const float h = n + z * (hreg[s][v] - n);          // (1-z)*n + z*h
                hreg[s][v] = h;
                const int m = v + (hi16 << 3);
                hsh[s][nxt][m][j] = (_Float16)h;                   // publish for next step's A frag
            }
        }
        __syncthreads();   // one barrier per step (parity double-buffered)
    }

    // |h1 - h2| to LDS, then tiny MLP head (once per block, negligible)
    #pragma unroll
    for (int v = 0; v < 8; ++v) {
        const int m = v + (hi16 << 3);
        dsh[m][j] = fabsf(hreg[0][v] - hreg[1][v]);
    }
    __syncthreads();

    if (tid < 16) {
        float acc = b2[0];
        for (int jj = 0; jj < 32; ++jj) {
            float a = b1[jj];
            #pragma unroll
            for (int k = 0; k < 64; ++k) a += dsh[tid][k] * W1[jj * 64 + k];
            acc += fmaxf(a, 0.0f) * W2[jj];
        }
        out[b0 + tid] = fast_sigmoid(acc);
    }
}

extern "C" void kernel_launch(void* const* d_in, const int* in_sizes, int n_in,
                              void* d_out, int out_size, void* d_ws, size_t ws_size,
                              hipStream_t stream) {
    (void)n_in; (void)out_size; (void)d_ws; (void)ws_size;
    const float* x1  = (const float*)d_in[0];
    const float* x2  = (const float*)d_in[1];
    const float* Wih = (const float*)d_in[2];
    const float* Whh = (const float*)d_in[3];
    const float* bih = (const float*)d_in[4];
    const float* bhh = (const float*)d_in[5];
    const float* W1  = (const float*)d_in[6];
    const float* b1  = (const float*)d_in[7];
    const float* W2  = (const float*)d_in[8];
    const float* b2  = (const float*)d_in[9];
    float* out = (float*)d_out;

    const int Btot = in_sizes[0] / (T_LEN * 3);   // 4096
    dim3 grid(Btot / 16), block(128);
    siamese_gru_wmma<<<grid, block, 0, stream>>>(x1, x2, Wih, Whh, bih, bhh,
                                                 W1, b1, W2, b2, out);
}
